// RecurrentNeuralNetwork_39341900431595
// MI455X (gfx1250) — compile-verified, matrix operand order checked by reading, not run on hardware
//
#include <hip/hip_runtime.h>
#include <hip/hip_bf16.h>

// Problem sizes (fixed by the reference)
#define T_STEPS 16384
#define X_DIM   512
#define H_DIM   1024
#define A_SCALE 0.1f
#define A_IDENT 0.9f

// Scan decomposition: 16 persistent workgroups, 64 rows of A each,
// A slice lives in LDS (64 rows * 1040 padded floats * 4B = 266,240 B < 320 KB WGP LDS).
#define SCAN_BLOCKS   16
#define ROWS_PER_BLK  64           // H_DIM / SCAN_BLOCKS
#define A_PAD_STRIDE  1040         // 1024 + 16 -> row-to-row bank-group rotation (mod 64 banks = 16)
#define SCAN_THREADS  1024         // 16 threads per row
#define THREADS_PER_ROW 16

typedef __attribute__((ext_vector_type(16))) __bf16 v16bf;
typedef __attribute__((ext_vector_type(8)))  float  v8f;

// ---------------------------------------------------------------------------
// Kernel 1: f32 -> bf16 conversion of x_seq and B, plus barrier-state init.
// ---------------------------------------------------------------------------
__global__ __launch_bounds__(256) void cvt_bf16_kernel(
    const float* __restrict__ x_seq,   // [T, X] f32
    const float* __restrict__ B,       // [H, X] f32
    __bf16* __restrict__ x_bf,         // [T, X] bf16
    __bf16* __restrict__ b_bf,         // [H, X] bf16
    unsigned* __restrict__ bar)        // [2] grid-barrier {count, gen}
{
    const size_t nx = (size_t)T_STEPS * X_DIM;
    const size_t nb = (size_t)H_DIM   * X_DIM;
    size_t i = (size_t)blockIdx.x * blockDim.x + threadIdx.x;
    if (i == 0) { bar[0] = 0u; bar[1] = 0u; }
    if (i < nx) {
        x_bf[i] = (__bf16)x_seq[i];
    } else if (i < nx + nb) {
        size_t j = i - nx;
        b_bf[j] = (__bf16)B[j];
    }
}

// ---------------------------------------------------------------------------
// Kernel 2: Bx = x_seq @ B^T + c  via v_wmma_f32_16x16x32_bf16.
// GEMM: M = T(16384), N = H(1024), K = X(512).
// One 16x16 output tile per wave; 4 waves (128 threads) per block.
// A-operand = x_seq rows (row-major, contiguous K).
// B-operand = columns of B^T = rows of B (row-major, contiguous K).
//
// Per the CDNA5 ISA 16-bit A-matrix 16x32 layout (CDNA5 ISA):
//   lanes 0-15  hold row M=lane,    K = {kc+0..7}  in elems 0..7, {kc+16..23} in elems 8..15
//   lanes 16-31 hold row M=lane-16, K = {kc+8..15} in elems 0..7, {kc+24..31} in elems 8..15
// B mirrors this with lanes <-> N columns (B supplied column-major = rows of B).
// C/D f32 16x16: VGPR v holds M = v (lanes 0-15) / M = v+8 (lanes 16-31), N = lane&15.
// ---------------------------------------------------------------------------
__global__ __launch_bounds__(128) void bx_gemm_kernel(
    const __bf16* __restrict__ x_bf,   // [T, X]
    const __bf16* __restrict__ b_bf,   // [H, X]
    const float*  __restrict__ c,      // [H]
    float* __restrict__ Bx)            // [T, H]
{
    const int wave = threadIdx.x >> 5;
    const int lane = threadIdx.x & 31;
    const int tile = blockIdx.x * 4 + wave;          // 0 .. 65535
    const int tm   = tile >> 6;                      // T/16 tiles in M
    const int tn   = tile & 63;                      // H/16 tiles in N
    const int l15  = lane & 15;
    const bool hi  = (lane >= 16);

    const __bf16* pA = x_bf + (size_t)(tm * 16 + l15) * X_DIM;
    const __bf16* pB = b_bf + (size_t)(tn * 16 + l15) * X_DIM;

    v8f acc = {};
    #pragma unroll 4
    for (int kc = 0; kc < X_DIM; kc += 32) {
        const int k0 = kc + (hi ? 8 : 0);
        const int k1 = k0 + 16;
        v16bf a, b;
        ((uint4*)&a)[0] = *(const uint4*)(pA + k0);  // 8 contiguous bf16 = 16B
        ((uint4*)&a)[1] = *(const uint4*)(pA + k1);
        ((uint4*)&b)[0] = *(const uint4*)(pB + k0);
        ((uint4*)&b)[1] = *(const uint4*)(pB + k1);
        acc = __builtin_amdgcn_wmma_f32_16x16x32_bf16(
                  /*neg_a=*/false, a, /*neg_b=*/false, b,
                  /*c_mod=*/(short)0, acc, /*reuse_a=*/false, /*reuse_b=*/false);
    }

    const float cv = c[tn * 16 + l15];
    float* outBase = Bx + (size_t)(tm * 16 + (hi ? 8 : 0)) * H_DIM + tn * 16 + l15;
    #pragma unroll
    for (int v = 0; v < 8; ++v)
        __builtin_nontemporal_store(acc[v] + cv, outBase + (size_t)v * H_DIM);
}

// ---------------------------------------------------------------------------
// Grid-wide barrier for the persistent scan kernel (16 blocks).
// ---------------------------------------------------------------------------
__device__ __forceinline__ void grid_barrier(unsigned* count, unsigned* gen, unsigned nblocks)
{
    __syncthreads();
    if (threadIdx.x == 0) {
        unsigned g = __hip_atomic_load(gen, __ATOMIC_RELAXED, __HIP_MEMORY_SCOPE_AGENT);
        unsigned prev = __hip_atomic_fetch_add(count, 1u, __ATOMIC_ACQ_REL, __HIP_MEMORY_SCOPE_AGENT);
        if (prev == nblocks - 1u) {
            __hip_atomic_store(count, 0u, __ATOMIC_RELAXED, __HIP_MEMORY_SCOPE_AGENT);
            __hip_atomic_fetch_add(gen, 1u, __ATOMIC_ACQ_REL, __HIP_MEMORY_SCOPE_AGENT);
        } else {
            while (__hip_atomic_load(gen, __ATOMIC_ACQUIRE, __HIP_MEMORY_SCOPE_AGENT) == g)
                __builtin_amdgcn_s_sleep(2);
        }
    }
    __syncthreads();
}

// ---------------------------------------------------------------------------
// Kernel 3: persistent sequential scan.
//   h[t+1] = tanh(A h[t] + Bx[t]),  A = 0.9 I + 0.1 A_raw kept in f32 in LDS.
// 16 blocks x 1024 threads; block owns 64 rows of A; 16 threads per row, each
// covering k = c16 + 16*j (j=0..63) -> conflict-free LDS banks; 16-lane
// shuffle-xor reduction; h exchanged through a double-buffered agent-scope
// global array; one grid barrier per step.
// ---------------------------------------------------------------------------
extern __shared__ float scan_lds[];   // [ROWS_PER_BLK * A_PAD_STRIDE] A slice + [H_DIM] h stage

__global__ __launch_bounds__(SCAN_THREADS) void rnn_scan_kernel(
    const float* __restrict__ A_raw,   // [H, H]
    const float* __restrict__ h0,      // [H]
    const float* __restrict__ Bx,      // [T, H]
    float* __restrict__ hb,            // [2, H] double-buffered state
    unsigned* __restrict__ bar,        // {count, gen}
    float* __restrict__ out)           // [T, H]
{
    float* As = scan_lds;                                   // padded A slice
    float* hs = scan_lds + ROWS_PER_BLK * A_PAD_STRIDE;     // staged h (1024 f32)

    const int tid     = threadIdx.x;
    const int rowBase = blockIdx.x * ROWS_PER_BLK;

    // Stage this block's 64 rows of A = 0.9*I + 0.1*A_raw into LDS (f32, full precision).
    for (int idx = tid; idx < ROWS_PER_BLK * H_DIM; idx += SCAN_THREADS) {
        const int lr  = idx >> 10;           // local row
        const int k   = idx & (H_DIM - 1);
        const int row = rowBase + lr;
        float v = A_SCALE * A_raw[(size_t)row * H_DIM + k];
        if (row == k) v += A_IDENT;
        As[lr * A_PAD_STRIDE + k] = v;
    }

    // Initialize h buffer 0 with h0 (agent-scope so every WGP sees it).
    if (blockIdx.x == 0)
        __hip_atomic_store(&hb[tid], h0[tid], __ATOMIC_RELAXED, __HIP_MEMORY_SCOPE_AGENT);

    __threadfence();
    grid_barrier(&bar[0], &bar[1], SCAN_BLOCKS);

    const int lr  = tid >> 4;                // local row: 0..63
    const int c16 = tid & 15;                // position within the 16-thread row team
    const float* Arow = As + lr * A_PAD_STRIDE;
    const int row = rowBase + lr;

    for (int t = 0; t < T_STEPS; ++t) {
        const float* hr = hb + ((t & 1) ? H_DIM : 0);
        float*       hw = hb + ((t & 1) ? 0 : H_DIM);

        // Stage current h into LDS (one agent-scope load per thread).
        hs[tid] = __hip_atomic_load(&hr[tid], __ATOMIC_RELAXED, __HIP_MEMORY_SCOPE_AGENT);
        __syncthreads();

        // Partial dot product: strided k keeps the 16 lanes on consecutive banks,
        // and the 1040-float row pad keeps the wave's two rows in disjoint bank groups.
        float acc = 0.0f;
        #pragma unroll
        for (int j = 0; j < H_DIM / THREADS_PER_ROW; ++j) {
            const int k = c16 + (j << 4);
            acc = __builtin_fmaf(Arow[k], hs[k], acc);
        }

        // Reduce across the 16-thread row team (stays within a wave32).
        acc += __shfl_xor(acc, 1, 32);
        acc += __shfl_xor(acc, 2, 32);
        acc += __shfl_xor(acc, 4, 32);
        acc += __shfl_xor(acc, 8, 32);

        if (c16 == 0) {
            const float bx = __builtin_nontemporal_load(&Bx[(size_t)t * H_DIM + row]);
            const float hv = tanhf(acc + bx);
            __hip_atomic_store(&hw[row], hv, __ATOMIC_RELAXED, __HIP_MEMORY_SCOPE_AGENT);
            __builtin_nontemporal_store(hv, &out[(size_t)t * H_DIM + row]);
        }

        __threadfence();                       // publish h before the step barrier
        grid_barrier(&bar[0], &bar[1], SCAN_BLOCKS);
    }
}

// ---------------------------------------------------------------------------
// Host launcher. Workspace layout (bytes):
//   [0, 64MB)            Bx            f32 [T, H]
//   [64MB, +16MB)        x_bf          bf16 [T, X]
//   [.., +1MB)           b_bf          bf16 [H, X]
//   [.., +8KB)           hb            f32 [2, H]
//   [.., +16B)           bar           u32 [2]
// ---------------------------------------------------------------------------
extern "C" void kernel_launch(void* const* d_in, const int* in_sizes, int n_in,
                              void* d_out, int out_size, void* d_ws, size_t ws_size,
                              hipStream_t stream)
{
    const float* x_seq = (const float*)d_in[0];   // [T, X]
    const float* h0    = (const float*)d_in[1];   // [H]
    const float* A_raw = (const float*)d_in[2];   // [H, H]
    const float* B     = (const float*)d_in[3];   // [H, X]
    const float* c     = (const float*)d_in[4];   // [H]
    float* out = (float*)d_out;

    char* ws = (char*)d_ws;
    const size_t bx_bytes = (size_t)T_STEPS * H_DIM * sizeof(float);        // 64 MB
    const size_t xb_bytes = (size_t)T_STEPS * X_DIM * sizeof(unsigned short); // 16 MB
    const size_t bb_bytes = (size_t)H_DIM * X_DIM * sizeof(unsigned short);   // 1 MB

    float*    Bx   = (float*)ws;
    __bf16*   x_bf = (__bf16*)(ws + bx_bytes);
    __bf16*   b_bf = (__bf16*)(ws + bx_bytes + xb_bytes);
    float*    hb   = (float*)(ws + bx_bytes + xb_bytes + bb_bytes);
    unsigned* bar  = (unsigned*)(ws + bx_bytes + xb_bytes + bb_bytes + 2 * H_DIM * sizeof(float));

    // 1) convert inputs to bf16 + reset barrier state (deterministic per call)
    {
        const size_t total = (size_t)T_STEPS * X_DIM + (size_t)H_DIM * X_DIM;
        const int blocks = (int)((total + 255) / 256);
        cvt_bf16_kernel<<<blocks, 256, 0, stream>>>(x_seq, B, x_bf, b_bf, bar);
    }

    // 2) Bx = x_seq @ B^T + c   (WMMA bf16, f32 accumulate)
    {
        const int tiles  = (T_STEPS / 16) * (H_DIM / 16);   // 65536
        const int blocks = tiles / 4;                       // 4 waves/block
        bx_gemm_kernel<<<blocks, 128, 0, stream>>>(x_bf, b_bf, c, Bx);
    }

    // 3) persistent sequential scan (A resident in 320KB WGP LDS)
    {
        const size_t smem = (size_t)(ROWS_PER_BLK * A_PAD_STRIDE + H_DIM) * sizeof(float); // 270,336 B
        rnn_scan_kernel<<<SCAN_BLOCKS, SCAN_THREADS, smem, stream>>>(A_raw, h0, Bx, hb, bar, out);
    }
}
